// MD_BiDirectional_Hamiltonian_56667798503980
// MI455X (gfx1250) — matrix operand order repaired and checked by reading, not compiled, not run on hardware
//
#include <hip/hip_runtime.h>
#include <math.h>

typedef __attribute__((ext_vector_type(16))) _Float16 v16h;
typedef __attribute__((ext_vector_type(8)))  _Float16 v8h;
typedef __attribute__((ext_vector_type(4)))  _Float16 v4h;
typedef __attribute__((ext_vector_type(8)))  float    v8f;

#define DT_STEP 0.1f

// ---------------------------------------------------------------------------
// Branchless tanh: gfx1250 hardware V_TANH_F32 (probe-confirmed lowering).
// ---------------------------------------------------------------------------
__device__ __forceinline__ float fast_tanh(float x) {
#if __has_builtin(__builtin_amdgcn_tanhf)
  return __builtin_amdgcn_tanhf(x);
#elif __has_builtin(__builtin_amdgcn_tanh_f32)
  return __builtin_amdgcn_tanh_f32(x);
#else
  float t;
#if __has_builtin(__builtin_amdgcn_exp2f)
  t = __builtin_amdgcn_exp2f(x * 2.8853900817779268f);
#else
  t = exp2f(x * 2.8853900817779268f);
#endif
#if __has_builtin(__builtin_amdgcn_rcpf)
  return 1.0f - 2.0f * __builtin_amdgcn_rcpf(t + 1.0f);
#else
  return 1.0f - 2.0f / (t + 1.0f);
#endif
#endif
}

// ---------------------------------------------------------------------------
// Zero-init of output (both directions accumulate with atomics).
// ---------------------------------------------------------------------------
__global__ void hnn_zero_kernel(float* __restrict__ out, int n) {
  int i = blockIdx.x * blockDim.x + threadIdx.x;
  if (i < n) out[i] = 0.0f;
}

// ---------------------------------------------------------------------------
// WMMA fragment helpers (wave32; A/B use contiguous k = 16*(lane/16) + e)
// ---------------------------------------------------------------------------

// A-matrix 16x32 f16 fragment; src row-major [16 x ld], 16B-aligned rows.
// lanes 0-15: M=lane, K=kbase+{0..15}; lanes 16-31: M=lane-16, K=kbase+{16..31}.
// Two aligned 16B vector loads -> ds_load_b128 x2.
__device__ __forceinline__ v16h load_a_frag(const _Float16* __restrict__ src,
                                            int ld, int kbase, int lane) {
  int m    = lane & 15;
  int koff = (lane >> 4) * 16;
  const _Float16* row = src + m * ld + kbase + koff;
  v8h lo = *(const v8h*)(row);
  v8h hi = *(const v8h*)(row + 8);
  return __builtin_shufflevector(lo, hi, 0, 1, 2, 3, 4, 5, 6, 7,
                                 8, 9, 10, 11, 12, 13, 14, 15);
}

// B-matrix 32x16 fragment from row-major W[K_total x ldw]: B[k][n]=W[k][n].
// Used only at init (hoisted to registers).
__device__ __forceinline__ v16h load_b_frag(const _Float16* __restrict__ W,
                                            int ldw, int kbase, int nbase, int lane) {
  int n  = nbase + (lane & 15);
  int k0 = kbase + (lane >> 4) * 16;
  v16h b;
#pragma unroll
  for (int e = 0; e < 16; ++e) b[e] = W[(k0 + e) * ldw + n];
  return b;
}

// Fragment of W^T: B[k][n] = W[n][k]; W row-major [N_total x ldw].
__device__ __forceinline__ v16h load_bT_frag(const _Float16* __restrict__ W,
                                             int ldw, int kbase, int nbase, int lane) {
  int n  = nbase + (lane & 15);
  int k0 = kbase + (lane >> 4) * 16;
  const _Float16* row = W + n * ldw + k0;
  v8h lo = *(const v8h*)(row);
  v8h hi = *(const v8h*)(row + 8);
  return __builtin_shufflevector(lo, hi, 0, 1, 2, 3, 4, 5, 6, 7,
                                 8, 9, 10, 11, 12, 13, 14, 15);
}

__device__ __forceinline__ v8f wmma_f16(v16h a, v16h b, v8f c) {
  return __builtin_amdgcn_wmma_f32_16x16x32_f16(
      /*neg_a=*/false, a, /*neg_b=*/false, b,
      /*c_mod=*/(short)0, c, /*reuse_a=*/false, /*reuse_b=*/false);
}

// Store this wave's 16x16 f32 tile as f16 into dst[16 x 128] at column nbase.
__device__ __forceinline__ void store_tile_f16(_Float16* __restrict__ dst,
                                               const float* v, int nbase, int lane) {
  int m0 = (lane >> 4) * 8;
  int n  = nbase + (lane & 15);
#pragma unroll
  for (int r = 0; r < 8; ++r) dst[(m0 + r) * 128 + n] = (_Float16)v[r];
}

// ---------------------------------------------------------------------------
// Main persistent-scan kernel.
// Block = 16 batch rows, one direction; 256 threads = 8 wave32.
// Wave w owns hidden columns [16w, 16w+16). All weight B-fragments live in
// registers for the whole scan; A-operands ping-pong between two LDS buffers
// so each stage costs exactly one barrier (6 barriers/step).
// ---------------------------------------------------------------------------
__launch_bounds__(256, 1)
__global__ void hnn_integrate_kernel(const float* __restrict__ x,
                                     const float* __restrict__ W1, const float* __restrict__ b1,
                                     const float* __restrict__ W2, const float* __restrict__ b2,
                                     const float* __restrict__ W3, const float* __restrict__ b3,
                                     const float* __restrict__ W4, const float* __restrict__ W5,
                                     float* __restrict__ out, int nsteps, int B) {
  __shared__ _Float16 sW1[64 * 128];    // 16 KB (fragment staging)
  __shared__ _Float16 sW2[128 * 128];   // 32 KB
  __shared__ _Float16 sW3[128 * 128];   // 32 KB
  __shared__ _Float16 sA0[16 * 128];    // 4 KB  ping
  __shared__ _Float16 sA1[16 * 128];    // 4 KB  pong
  __shared__ float    sY[16 * 64];      // 4 KB  f32 state (master copy)
  __shared__ float    sB1[128], sB2[128], sB3[128], sV4[128];

  const int tid     = threadIdx.x;
  const int nTiles  = B >> 4;
  const bool fwd    = (blockIdx.x < nTiles);
  const int tile    = fwd ? blockIdx.x : (blockIdx.x - nTiles);
  const int r0      = tile * 16;
  const int wv      = tid >> 5;
  const int lane    = tid & 31;
  const int nb      = wv * 16;           // this wave's N-tile base column
  const int nloc    = nb + (lane & 15);  // this lane's global column
  const int total_j = nsteps + 1;

  // ---- stage weights into LDS (f16), biases, v4 = W4 @ W5 ----
  for (int i = tid; i < 64 * 128; i += 256) sW1[i] = (_Float16)W1[i];
  for (int i = tid; i < 128 * 128; i += 256) sW2[i] = (_Float16)W2[i];
  for (int i = tid; i < 128 * 128; i += 256) sW3[i] = (_Float16)W3[i];
  if (tid < 128) {
    sB1[tid] = b1[tid]; sB2[tid] = b2[tid]; sB3[tid] = b3[tid];
    float s = 0.0f;
    for (int k = 0; k < 128; ++k) s += W4[tid * 128 + k] * W5[k];
    sV4[tid] = s;
  }
  // ---- initial state (backward: x2 * vflip); f32 master + f16 A-copy ----
  for (int i = tid; i < 16 * 64; i += 256) {
    int m = i >> 6, f = i & 63;
    float v = x[((size_t)(r0 + m) * 2 + (fwd ? 0 : 1)) * 64 + f];
    if (!fwd && f >= 32) v = -v;
    sY[i]  = v;
    sA0[i] = (_Float16)v;   // 16x64, ld=64
  }
  __syncthreads();

  // ---- hoist all loop-invariant weight B-fragments into registers ----
  const int nbW1 = (wv < 4) ? nb : 0;   // W1^T has only 64 output columns
  v16h fW1[2], fW2[4], fW3[4], fW3T[4], fW2T[4], fW1T[4];
#pragma unroll
  for (int c = 0; c < 2; ++c) fW1[c] = load_b_frag(sW1, 128, c * 32, nb, lane);
#pragma unroll
  for (int c = 0; c < 4; ++c) {
    fW2[c]  = load_b_frag(sW2, 128, c * 32, nb, lane);
    fW3[c]  = load_b_frag(sW3, 128, c * 32, nb, lane);
    fW3T[c] = load_bT_frag(sW3, 128, c * 32, nb, lane);
    fW2T[c] = load_bT_frag(sW2, 128, c * 32, nb, lane);
    fW1T[c] = load_bT_frag(sW1, 128, c * 32, nbW1, lane);
  }
  const float biasZ1 = sB1[nloc];
  const float biasZ2 = sB2[nloc];
  const float biasZ3 = sB3[nloc];
  const float v4v    = sV4[nloc];

  // ---- store contribution of raw step index 0 (weight 1.0) ----
  {
    int j = fwd ? 0 : nsteps;
    for (int i = tid; i < 16 * 64; i += 256) {
      int m = i >> 6, f = i & 63;
      float v = sY[i];
      if (!fwd && f >= 32) v = -v;   // un-flip for output
      unsafeAtomicAdd(out + ((size_t)(r0 + m) * total_j + j) * 64 + f, v);
    }
  }
  __syncthreads();

  // ---- time scan: steps 1 .. nsteps-1 (step nsteps has weight 0) ----
  for (int s = 1; s < nsteps; ++s) {
    float h1r[8], h2r[8], dv[8];

    // ---- L1: h1 = tanh(y @ W1 + b1)          [reads sA0(ld64) -> sA1] ----
    {
      v8f acc;
#pragma unroll
      for (int r = 0; r < 8; ++r) acc[r] = biasZ1;
      acc = wmma_f16(load_a_frag(sA0, 64, 0, lane), fW1[0], acc);
      acc = wmma_f16(load_a_frag(sA0, 64, 32, lane), fW1[1], acc);
#pragma unroll
      for (int r = 0; r < 8; ++r) h1r[r] = fast_tanh(acc[r]);
      store_tile_f16(sA1, h1r, nb, lane);
    }
    __syncthreads();

    // ---- L2: h2 = tanh(h1 @ W2 + b2)         [reads sA1 -> sA0] ----
    {
      v8f acc;
#pragma unroll
      for (int r = 0; r < 8; ++r) acc[r] = biasZ2;
#pragma unroll
      for (int c = 0; c < 4; ++c)
        acc = wmma_f16(load_a_frag(sA1, 128, c * 32, lane), fW2[c], acc);
#pragma unroll
      for (int r = 0; r < 8; ++r) h2r[r] = fast_tanh(acc[r]);
      store_tile_f16(sA0, h2r, nb, lane);
    }
    __syncthreads();

    // ---- L3: dz3 = v4 * (1 - tanh(z3)^2)     [reads sA0 -> sA1] ----
    {
      v8f acc;
#pragma unroll
      for (int r = 0; r < 8; ++r) acc[r] = biasZ3;
#pragma unroll
      for (int c = 0; c < 4; ++c)
        acc = wmma_f16(load_a_frag(sA0, 128, c * 32, lane), fW3[c], acc);
#pragma unroll
      for (int r = 0; r < 8; ++r) {
        float h3 = fast_tanh(acc[r]);
        dv[r] = v4v * (1.0f - h3 * h3);
      }
      store_tile_f16(sA1, dv, nb, lane);
    }
    __syncthreads();

    // ---- B3: dz2 = (dz3 @ W3^T)*(1-h2^2)     [reads sA1 -> sA0] ----
    {
      v8f acc = {};
#pragma unroll
      for (int c = 0; c < 4; ++c)
        acc = wmma_f16(load_a_frag(sA1, 128, c * 32, lane), fW3T[c], acc);
#pragma unroll
      for (int r = 0; r < 8; ++r) dv[r] = acc[r] * (1.0f - h2r[r] * h2r[r]);
      store_tile_f16(sA0, dv, nb, lane);
    }
    __syncthreads();

    // ---- B2: dz1 = (dz2 @ W2^T)*(1-h1^2)     [reads sA0 -> sA1] ----
    {
      v8f acc = {};
#pragma unroll
      for (int c = 0; c < 4; ++c)
        acc = wmma_f16(load_a_frag(sA0, 128, c * 32, lane), fW2T[c], acc);
#pragma unroll
      for (int r = 0; r < 8; ++r) dv[r] = acc[r] * (1.0f - h1r[r] * h1r[r]);
      store_tile_f16(sA1, dv, nb, lane);
    }
    __syncthreads();

    // ---- B1: g = dz1 @ W1^T; symplectic update of state ----
    //      [reads sA1 -> writes sY (f32) AND sA0 (f16 next A-operand)]
    if (wv < 4) {
      v8f acc = {};
#pragma unroll
      for (int c = 0; c < 4; ++c)
        acc = wmma_f16(load_a_frag(sA1, 128, c * 32, lane), fW1T[c], acc);
      // dyn = [g_p, -g_q]: y[:, n-32] += DT*g[n] (n>=32); y[:, n+32] -= DT*g[n] (n<32)
      int m0 = (lane >> 4) * 8;
      int n  = nb + (lane & 15);
      int tgt = (n < 32) ? (n + 32) : (n - 32);
      float sgn = (n < 32) ? -DT_STEP : DT_STEP;
#pragma unroll
      for (int r = 0; r < 8; ++r) {
        int idx = (m0 + r) * 64 + tgt;
        float nv = sY[idx] + sgn * acc[r];
        sY[idx]  = nv;
        sA0[idx] = (_Float16)nv;   // next step's A-operand, no cvt pass needed
      }
    }
    __syncthreads();

    // ---- accumulate weighted trajectory sample (no trailing barrier) ----
    {
      float scale = (float)(nsteps - s) / (float)nsteps;  // == w0[j] fwd, w1[j] bwd
      int j = fwd ? s : (nsteps - s);
      for (int i = tid; i < 16 * 64; i += 256) {
        int m = i >> 6, f = i & 63;
        float v = sY[i] * scale;
        if (!fwd && f >= 32) v = -v;  // un-flip
        unsafeAtomicAdd(out + ((size_t)(r0 + m) * total_j + j) * 64 + f, v);
      }
    }
  }
}

// ---------------------------------------------------------------------------
// Entry point
// ---------------------------------------------------------------------------
extern "C" void kernel_launch(void* const* d_in, const int* in_sizes, int n_in,
                              void* d_out, int out_size, void* d_ws, size_t ws_size,
                              hipStream_t stream) {
  // inputs: t, x, W1,b1, W2,b2, W3,b3, W4,b4, W5,b5
  const float* x  = (const float*)d_in[1];
  const float* W1 = (const float*)d_in[2];
  const float* b1 = (const float*)d_in[3];
  const float* W2 = (const float*)d_in[4];
  const float* b2 = (const float*)d_in[5];
  const float* W3 = (const float*)d_in[6];
  const float* b3 = (const float*)d_in[7];
  const float* W4 = (const float*)d_in[8];
  const float* W5 = (const float*)d_in[10];
  float* out = (float*)d_out;

  const int F = 64;
  const int B = in_sizes[1] / (2 * F);          // x is (B, 2, F)
  const int total_len = out_size / (B * F);     // t + 2
  const int nsteps = total_len - 1;             // t + 1

  hnn_zero_kernel<<<(out_size + 255) / 256, 256, 0, stream>>>(out, out_size);

  const int nTiles = B / 16;
  hnn_integrate_kernel<<<nTiles * 2, 256, 0, stream>>>(
      x, W1, b1, W2, b2, W3, b3, W4, W5, out, nsteps, B);
}
